// LSTMGCNModel_89979564851474
// MI455X (gfx1250) — matrix operand order repaired and checked by reading, not compiled, not run on hardware
//
#include <hip/hip_runtime.h>
#include <hip/hip_bf16.h>

// ---------------------------------------------------------------------------
// LSTM-GCN forward, MI455X (gfx1250, wave32).
//
// The GCN branch of the reference is dead code (result discarded), so all
// edge/segment-sum work is skipped. Only the temporal path survives:
// scalar->16 affine, 2x LSTM(hid=32, T=12), fc1(relu), fc2. The LSTM-0 input
// projection folds to rank-1: pre_x0 = s * u + v with
//   u = Wih0 @ W_tp[0],  v = Wih0 @ b_tp + bih0 + bhh0.
//
// One wave32 per 16-node tile. All matmuls are v_wmma_f32_16x16x32_f16:
// A = weight tile [16 gates x 32 hid] (f16, pre-packed in A VGPR layout,
// staged in LDS and register-promoted by the compiler), B = hidden state
// [32 hid x 16 nodes] (f16, rebuilt each step via cvt_pk_rtz + half-wave
// shfl_xor), C/D = gate tile [16 x 16] f32 whose layout matches the
// cell-state layout, so nonlinearities are straight per-lane VALU.
//
// The step loop stays ROLLED (full unroll caused scratch spills). All 12
// temporal scalars are loaded ONCE before the loop into three f4v registers;
// the per-step value is picked with a wave-uniform select tree, so the
// recurrence loop body contains no global loads and no load waits.
// ---------------------------------------------------------------------------

typedef __attribute__((ext_vector_type(16))) _Float16 v16h;
typedef __attribute__((ext_vector_type(2)))  __fp16   pk2h;   // cvt_pkrtz ret
typedef __attribute__((ext_vector_type(8)))  float    v8f;
typedef __attribute__((ext_vector_type(4)))  float    f4v;

#define NUM_FEATS   32
#define SEQ_LEN     12
#define N_WTILES    26            // 8 Whh0, 8 Wih1, 8 Whh1, 1 fc1, 1 fc2
#define LANE_STRIDE 12            // dwords per lane in a tile (8 used + pad)
#define TILE_STRIDE (32 * LANE_STRIDE)          // 384 dwords per tile
#define A_REGION    (N_WTILES * TILE_STRIDE)    // 9984 dwords
#define U_OFF       A_REGION                    // u[128] f32
#define V_OFF       (U_OFF + 128)               // v[128] f32
#define B1C_OFF     (V_OFF + 128)               // bih1+bhh1 [128] f32
#define BFC1_OFF    (B1C_OFF + 128)             // b_fc1 [16] f32
#define WS_DWORDS   (BFC1_OFF + 16)             // 10384 dwords = 41.5 KB

// ---------------------------------------------------------------------------
// small device helpers
// ---------------------------------------------------------------------------
__device__ inline float fast_sig(float x) {
  float e = __builtin_amdgcn_exp2f(x * -1.44269504f);
  return __builtin_amdgcn_rcpf(1.0f + e);
}

__device__ inline float fast_tanh(float x) {
#if __has_builtin(__builtin_amdgcn_tanhf)
  return __builtin_amdgcn_tanhf(x);
#elif __has_builtin(__builtin_amdgcn_tanh_f32)
  return __builtin_amdgcn_tanh_f32(x);
#else
  float e = __builtin_amdgcn_exp2f(x * 2.88539008f);
  return 1.0f - 2.0f * __builtin_amdgcn_rcpf(e + 1.0f);
#endif
}

__device__ inline unsigned pk16(float a, float b) {
  union { pk2h h; unsigned u; } cv;
  cv.h = __builtin_amdgcn_cvt_pkrtz(a, b);   // v_cvt_pk_rtz_f16_f32
  return cv.u;
}

__device__ inline v8f wmma_f16(v16h a, v16h b, v8f c) {
  return __builtin_amdgcn_wmma_f32_16x16x32_f16(
      /*neg_a=*/false, a, /*neg_b=*/false, b,
      /*c_mod=*/(short)0, c, /*reuse_a=*/false, /*reuse_b=*/false);
}

// Load one pre-packed A-tile (16x32 f16, WMMA A layout) from LDS.
__device__ inline v16h load_tileA(const unsigned* lds, int t, int lane) {
  union { f4v q[2]; v16h v; } a;
  const float* p = (const float*)(lds + t * TILE_STRIDE + lane * LANE_STRIDE);
  a.q[0] = *(const f4v*)p;
  a.q[1] = *(const f4v*)(p + 4);
  return a.v;
}

// D-layout f32 pair (hidden 0-15, 16-31) -> WMMA B layout f16 [K=32 x N=16].
// Lane (n, hi): d0[j] = h[8*hi + j], d1[j] = h[16 + 8*hi + j].
// B (contiguous k order, as A packing) gets VGPR v: halves h[16*hi + 2v], h[16*hi + 2v + 1].
__device__ inline v16h relayout_hB(v8f d0, v8f d1, int hi) {
  union { unsigned u[8]; v16h v; } b;
#pragma unroll
  for (int jj = 0; jj < 4; ++jj) {
    unsigned p0 = pk16(d0[2 * jj], d0[2 * jj + 1]);
    unsigned p1 = pk16(d1[2 * jj], d1[2 * jj + 1]);
    unsigned t0 = (unsigned)__shfl_xor((int)p0, 16, 32);
    unsigned t1 = (unsigned)__shfl_xor((int)p1, 16, 32);
    b.u[jj]     = hi ? t1 : p0;
    b.u[4 + jj] = hi ? p1 : t0;
  }
  return b.v;
}

// 16-row B (K=0..15 live, K=16..31 zero) from a single D-layout tile.
__device__ inline v16h relayout_B16(v8f d, int hi) {
  union { unsigned u[8]; v16h v; } b;
#pragma unroll
  for (int jj = 0; jj < 4; ++jj) {
    unsigned p = pk16(d[2 * jj], d[2 * jj + 1]);
    unsigned t = (unsigned)__shfl_xor((int)p, 16, 32);
    b.u[jj]     = hi ? 0u : p;
    b.u[4 + jj] = hi ? 0u : t;
  }
  return b.v;
}

__device__ inline v8f mk8(f4v a, f4v b) {
  v8f r;
  r[0] = a.x; r[1] = a.y; r[2] = a.z; r[3] = a.w;
  r[4] = b.x; r[5] = b.y; r[6] = b.z; r[7] = b.w;
  return r;
}

__device__ inline v8f fma8(float s, f4v u0, f4v u1, f4v w0, f4v w1) {
  v8f r;
  r[0] = fmaf(s, u0.x, w0.x); r[1] = fmaf(s, u0.y, w0.y);
  r[2] = fmaf(s, u0.z, w0.z); r[3] = fmaf(s, u0.w, w0.w);
  r[4] = fmaf(s, u1.x, w1.x); r[5] = fmaf(s, u1.y, w1.y);
  r[6] = fmaf(s, u1.z, w1.z); r[7] = fmaf(s, u1.w, w1.w);
  return r;
}

// Pick element (st) out of 12 register-resident floats with a wave-uniform
// select tree (st is an SGPR value: 2 uniform branches + 3 cndmasks).
__device__ inline float pick12(int st, f4v A, f4v B, f4v C) {
  f4v q = (st < 4) ? A : ((st < 8) ? B : C);
  int r = st & 3;
  float lo = (r & 1) ? q.y : q.x;
  float hh = (r & 1) ? q.w : q.z;
  return (r & 2) ? hh : lo;
}

// PyTorch gate order i,f,g,o -> gate tiles {0,1}{2,3}{4,5}{6,7}; tile-pair
// layout matches the c-state layout exactly, so this is pure per-lane math.
__device__ inline void lstm_cell(const v8f* g, v8f& ca, v8f& cb,
                                 v8f& hd0, v8f& hd1) {
#pragma unroll
  for (int j = 0; j < 8; ++j) {
    float ig = fast_sig(g[0][j]);
    float fg = fast_sig(g[2][j]);
    float gg = fast_tanh(g[4][j]);
    float og = fast_sig(g[6][j]);
    ca[j] = fg * ca[j] + ig * gg;
    hd0[j] = og * fast_tanh(ca[j]);

    ig = fast_sig(g[1][j]);
    fg = fast_sig(g[3][j]);
    gg = fast_tanh(g[5][j]);
    og = fast_sig(g[7][j]);
    cb[j] = fg * cb[j] + ig * gg;
    hd1[j] = og * fast_tanh(cb[j]);
  }
}

// ---------------------------------------------------------------------------
// Kernel 1: pack weights into WMMA A-layout f16 + folded constants, into d_ws.
// Tile t, lane l (m=l&15, hi=l>>4), dword d: halves (k0, k0+1) with
//   k0 = 16*hi + 2d        (contiguous k order, matches B relayout)
// ---------------------------------------------------------------------------
__global__ void lstm_pack(const float* __restrict__ W_tp,
                          const float* __restrict__ b_tp,
                          const float* __restrict__ Wih0,
                          const float* __restrict__ Whh0,
                          const float* __restrict__ bih0,
                          const float* __restrict__ bhh0,
                          const float* __restrict__ Wih1,
                          const float* __restrict__ Whh1,
                          const float* __restrict__ bih1,
                          const float* __restrict__ bhh1,
                          const float* __restrict__ W_fc1,
                          const float* __restrict__ b_fc1,
                          const float* __restrict__ W_fc2,
                          unsigned* __restrict__ ws) {
  int tid = blockIdx.x * blockDim.x + threadIdx.x;
  int nth = gridDim.x * blockDim.x;

  // A tiles (packed f16 pairs)
  for (int i = tid; i < N_WTILES * 32 * 8; i += nth) {
    int t = i >> 8, r = i & 255;
    int lane = r >> 3, d = r & 7;
    int m = lane & 15, hi = lane >> 4;
    int k0 = 16 * hi + 2 * d;
    float a0 = 0.0f, a1 = 0.0f;
    if (t < 8)       { int g = t * 16 + m;        a0 = Whh0[g * 32 + k0]; a1 = Whh0[g * 32 + k0 + 1]; }
    else if (t < 16) { int g = (t - 8) * 16 + m;  a0 = Wih1[g * 32 + k0]; a1 = Wih1[g * 32 + k0 + 1]; }
    else if (t < 24) { int g = (t - 16) * 16 + m; a0 = Whh1[g * 32 + k0]; a1 = Whh1[g * 32 + k0 + 1]; }
    else if (t == 24){ a0 = W_fc1[k0 * 16 + m];   a1 = W_fc1[(k0 + 1) * 16 + m]; }          // fc1^T
    else             { if (m == 0 && k0 < 16) { a0 = W_fc2[k0]; a1 = W_fc2[k0 + 1]; } }     // fc2 row 0
    union { _Float16 h[2]; unsigned u; } cv;
    cv.h[0] = (_Float16)a0; cv.h[1] = (_Float16)a1;
    ws[t * TILE_STRIDE + lane * LANE_STRIDE + d] = cv.u;
  }
  // zero the pad dwords (d_ws is poison-initialized)
  for (int i = tid; i < N_WTILES * 32 * 4; i += nth) {
    int t = i >> 7, r = i & 127;
    int lane = r >> 2, d = r & 3;
    ws[t * TILE_STRIDE + lane * LANE_STRIDE + 8 + d] = 0u;
  }
  // folded constants: u, v (LSTM0 rank-1 input), b1c, b_fc1
  float* wf = (float*)ws;
  for (int i = tid; i < 128; i += nth) {
    float u = 0.0f, v = 0.0f;
    for (int f = 0; f < 16; ++f) {
      float w = Wih0[i * 16 + f];
      u += W_tp[f] * w;
      v += b_tp[f] * w;
    }
    v += bih0[i] + bhh0[i];
    wf[U_OFF + i]   = u;
    wf[V_OFF + i]   = v;
    wf[B1C_OFF + i] = bih1[i] + bhh1[i];
  }
  for (int i = tid; i < 16; i += nth) wf[BFC1_OFF + i] = b_fc1[i];
}

// ---------------------------------------------------------------------------
// Kernel 2: the fused temporal model. One wave32 per 16-node tile.
// ---------------------------------------------------------------------------
__global__ __launch_bounds__(256) void lstm_main(
    const float* __restrict__ x, const unsigned* __restrict__ ws,
    const float* __restrict__ bfc2p, float* __restrict__ out,
    int n_nodes, int n_tiles) {
  __shared__ unsigned lds[WS_DWORDS];
  int tid = threadIdx.x;

  int wave = tid >> 5;
  int lane = tid & 31;
  int tile_id = blockIdx.x * 8 + wave;
  int hi = lane >> 4;
  int node = tile_id * 16 + (lane & 15);
  int nodeL = node < n_nodes ? node : n_nodes - 1;   // load clamp only
  const float* sp = x + (size_t)nodeL * NUM_FEATS + (NUM_FEATS - SEQ_LEN);

  // Issue the temporal-scalar loads first; their latency overlaps the
  // LDS stage-in + barrier below.
  f4v sA = *(const f4v*)(sp + 0);
  f4v sB = *(const f4v*)(sp + 4);
  f4v sC = *(const f4v*)(sp + 8);

  for (int i = tid; i < WS_DWORDS; i += 256) lds[i] = ws[i];
  __syncthreads();
  if (tile_id >= n_tiles) return;          // wave-uniform exit

  const float* cf = (const float*)lds;
  float bfc2 = bfc2p[0];

  v16h h0B = {};         // hidden states in WMMA B layout (f16)
  v16h h1B = {};
  v8f c0a = {}, c0b = {};  // cell states in D layout (f32)
  v8f c1a = {}, c1b = {};

#pragma unroll 1
  for (int st = 0; st < SEQ_LEN; ++st) {
    float sv = pick12(st, sA, sB, sC);   // register-resident, no memory op

    v8f g[8];
    v8f hd0, hd1;

    // ---- Layer 0: C-init = s*u + v (rank-1 folded input), then Whh0 @ h0
#pragma unroll
    for (int t = 0; t < 8; ++t) {
      const float* up = cf + U_OFF + 16 * t + 8 * hi;
      const float* vp = cf + V_OFF + 16 * t + 8 * hi;
      f4v u0 = *(const f4v*)up, u1 = *(const f4v*)(up + 4);
      f4v w0 = *(const f4v*)vp, w1 = *(const f4v*)(vp + 4);
      g[t] = fma8(sv, u0, u1, w0, w1);
    }
#pragma unroll
    for (int t = 0; t < 8; ++t)
      g[t] = wmma_f16(load_tileA(lds, t, lane), h0B, g[t]);
    lstm_cell(g, c0a, c0b, hd0, hd1);
    h0B = relayout_hB(hd0, hd1, hi);

    // ---- Layer 1: C-init = bih1+bhh1, then Wih1 @ h0_new + Whh1 @ h1
#pragma unroll
    for (int t = 0; t < 8; ++t) {
      const float* bp = cf + B1C_OFF + 16 * t + 8 * hi;
      v8f c = mk8(*(const f4v*)bp, *(const f4v*)(bp + 4));
      c = wmma_f16(load_tileA(lds, 8 + t, lane), h0B, c);
      g[t] = wmma_f16(load_tileA(lds, 16 + t, lane), h1B, c);
    }
    lstm_cell(g, c1a, c1b, hd0, hd1);
    h1B = relayout_hB(hd0, hd1, hi);
  }

  // ---- fc1(relu) + fc2 via two more WMMAs
  const float* bp = cf + BFC1_OFF + 8 * hi;
  v8f z = mk8(*(const f4v*)bp, *(const f4v*)(bp + 4));
  z = wmma_f16(load_tileA(lds, 24, lane), h1B, z);
#pragma unroll
  for (int j = 0; j < 8; ++j) z[j] = fmaxf(z[j], 0.0f);
  v16h fB = relayout_B16(z, hi);
  v8f o = {};
  o = wmma_f16(load_tileA(lds, 25, lane), fB, o);

  // D row M=0 lives in VGPR0 of lanes 0..15
  if (lane < 16 && node < n_nodes) out[node] = o[0] + bfc2;
}

// ---------------------------------------------------------------------------
// host launcher
// ---------------------------------------------------------------------------
extern "C" void kernel_launch(void* const* d_in, const int* in_sizes, int n_in,
                              void* d_out, int out_size, void* d_ws, size_t ws_size,
                              hipStream_t stream) {
  const float* x     = (const float*)d_in[0];
  // GCN inputs d_in[1..9] are dead code in the reference -> ignored.
  const float* W_tp  = (const float*)d_in[10];
  const float* b_tp  = (const float*)d_in[11];
  const float* Wih0  = (const float*)d_in[12];
  const float* Whh0  = (const float*)d_in[13];
  const float* bih0  = (const float*)d_in[14];
  const float* bhh0  = (const float*)d_in[15];
  const float* Wih1  = (const float*)d_in[16];
  const float* Whh1  = (const float*)d_in[17];
  const float* bih1  = (const float*)d_in[18];
  const float* bhh1  = (const float*)d_in[19];
  const float* W_fc1 = (const float*)d_in[20];
  const float* b_fc1 = (const float*)d_in[21];
  const float* W_fc2 = (const float*)d_in[22];
  const float* b_fc2 = (const float*)d_in[23];
  unsigned* ws = (unsigned*)d_ws;

  int N = in_sizes[0] / NUM_FEATS;
  int n_tiles = (N + 15) / 16;

  lstm_pack<<<32, 256, 0, stream>>>(W_tp, b_tp, Wih0, Whh0, bih0, bhh0,
                                    Wih1, Whh1, bih1, bhh1,
                                    W_fc1, b_fc1, W_fc2, ws);

  int blocks = (n_tiles + 7) / 8;   // 8 waves per 256-thread block
  lstm_main<<<blocks, 256, 0, stream>>>(x, ws, b_fc2, (float*)d_out,
                                        N, n_tiles);
}
